// ProteinStructureTransformer_77841987272942
// MI455X (gfx1250) — matrix operand order repaired and checked by reading, not compiled, not run on hardware
//
#include <hip/hip_runtime.h>
#include <hip/hip_bf16.h>
#include <math.h>

// ---------- CDNA5 WMMA types ----------
typedef __attribute__((ext_vector_type(16))) _Float16 v16h;
typedef __attribute__((ext_vector_type(8)))  _Float16 v8h;
typedef __attribute__((ext_vector_type(8)))  float    v8f;
typedef __attribute__((ext_vector_type(4)))  unsigned int su32x4;
typedef __attribute__((ext_vector_type(8)))  unsigned int su32x8;

#define SEQ 384
#define DIM 1024
#define NL  12
#define NH  16
#define HD  64
#define DFF 4096

// =====================================================================
// Tensor Data Mover: DMA a 2D tile (tile_d0 x tile_d1 elements, 2B each,
// row stride stride_d0 elements) from global memory into LDS.
// D# built per CDNA5 ISA 8.3/8.4; issued via inline asm (toolchain-portable).
// Completion tracked with TENSORcnt (s_wait_tensorcnt).
// =====================================================================
__device__ __forceinline__ void tdm_load_2d_f16(unsigned lds_off, const void* gptr,
                                                unsigned tile_d0, unsigned tile_d1,
                                                unsigned stride_d0) {
  const unsigned long long ga = (unsigned long long)gptr;
  su32x4 g0;
  g0[0] = 1u;                                   // count=1 (valid descriptor)
  g0[1] = lds_off;                              // lds_addr
  g0[2] = (unsigned)ga;                         // global_addr[31:0]
  g0[3] = ((unsigned)(ga >> 32) & 0x01FFFFFFu)  // global_addr[56:32]
          | 0x80000000u;                        // type=2 ("image")
  su32x8 g1;
  g1[0] = 0x00010000u;                          // workgroup_mask=0, data_size=1 (2B)
  g1[1] = tile_d0 << 16;                        // tensor_dim0[15:0]
  g1[2] = tile_d1 << 16;                        // tensor_dim1[15:0]
  g1[3] = tile_d0 << 16;                        // tile_dim0
  g1[4] = tile_d1;                              // tile_dim1 (tile_dim2=0)
  g1[5] = stride_d0;                            // tensor_dim0_stride[31:0]
  g1[6] = 0u;
  g1[7] = 0u;
  su32x4 g2 = {0u, 0u, 0u, 0u};
  su32x4 g3 = {0u, 0u, 0u, 0u};
  asm volatile("tensor_load_to_lds %0, %1, %2, %3"
               :: "s"(g0), "s"(g1), "s"(g2), "s"(g3) : "memory");
}

// =====================================================================
// Embedding: x[s,:] = emb[seq[s],:] + pos[0,s,:]   (writes f32 + f16)
// =====================================================================
__global__ void k_embed(const int* __restrict__ seq, const float* __restrict__ emb,
                        const float* __restrict__ pos,
                        float* __restrict__ x32, _Float16* __restrict__ x16) {
  const int s = blockIdx.x;
  const int c = threadIdx.x * 4;
  const int tok = seq[s];
  const float4 e = *(const float4*)(emb + (size_t)tok * DIM + c);
  const float4 p = *(const float4*)(pos + (size_t)s * DIM + c);
  float v[4] = {e.x + p.x, e.y + p.y, e.z + p.z, e.w + p.w};
  float* o32 = x32 + (size_t)s * DIM + c;
  _Float16* o16 = x16 + (size_t)s * DIM + c;
#pragma unroll
  for (int t = 0; t < 4; ++t) { o32[t] = v[t]; o16[t] = (_Float16)v[t]; }
}

// =====================================================================
// f32 -> f16 conversion (weight staging)
// =====================================================================
__global__ void k_cvt(const float* __restrict__ src, _Float16* __restrict__ dst, int n) {
  int i = blockIdx.x * 1024 + threadIdx.x * 4;
  if (i + 3 < n) {
    float4 v = *(const float4*)(src + i);
    dst[i + 0] = (_Float16)v.x; dst[i + 1] = (_Float16)v.y;
    dst[i + 2] = (_Float16)v.z; dst[i + 3] = (_Float16)v.w;
  }
}

// =====================================================================
// A-fragment loader, contiguous k order (k = 16*half + s), matching the B loads
// =====================================================================
__device__ __forceinline__ v16h load_a_tile(const _Float16* Arow, int k, int half_sel) {
  v8h lo = *(const v8h*)(Arow + k + half_sel * 16);
  v8h hi = *(const v8h*)(Arow + k + half_sel * 16 + 8);
  v16h a;
#pragma unroll
  for (int t = 0; t < 8; ++t) { a[t] = lo[t]; a[8 + t] = hi[t]; }
  return a;
}

// =====================================================================
// NT WMMA GEMM, 16x64 tile per wave (4 N-tiles share one A fragment):
// C[M,N] = act(A16[M,K](lda) * B16[N,K](ldb)^T + bias). ACT:0 none 1 gelu 2 relu
// =====================================================================
template <int ACT, bool W32, bool W16>
__global__ void k_gemm_nt4(const _Float16* __restrict__ A, int lda,
                           const _Float16* __restrict__ B, int ldb,
                           const float* __restrict__ bias,
                           float* __restrict__ C32, _Float16* __restrict__ C16,
                           int ldc, int M, int N, int K) {
  const int lane = threadIdx.x & 31;
  const int hs = lane >> 4;     // 0 or 1
  const int l16 = lane & 15;
  const int m0 = blockIdx.y * 16;
  const int n0 = blockIdx.x * 64;
  const _Float16* Arow = A + (size_t)(m0 + l16) * lda;
  const _Float16* Br[4];
#pragma unroll
  for (int t = 0; t < 4; ++t) Br[t] = B + (size_t)(n0 + t * 16 + l16) * ldb;

  v8f acc[4];
#pragma unroll
  for (int t = 0; t < 4; ++t) acc[t] = (v8f){};

  for (int k = 0; k < K; k += 32) {
    if (k + 256 < K) {
      __builtin_prefetch(Br[0] + k + 256, 0, 0);   // global_prefetch_b8
      __builtin_prefetch(Br[2] + k + 256, 0, 0);
    }
    v16h a = load_a_tile(Arow, k, hs);
#pragma unroll
    for (int t = 0; t < 4; ++t) {
      v16h b = *(const v16h*)(Br[t] + k + hs * 16);
      acc[t] = __builtin_amdgcn_wmma_f32_16x16x32_f16(false, a, false, b, (short)0,
                                                      acc[t], false, false);
    }
  }
#pragma unroll
  for (int t = 0; t < 4; ++t) {
    const int n = n0 + t * 16 + l16;
    const float bv = bias ? bias[n] : 0.0f;
#pragma unroll
    for (int r = 0; r < 8; ++r) {
      const int m = m0 + r + hs * 8;
      float v = acc[t][r] + bv;
      if (ACT == 1) v = 0.5f * v * (1.0f + erff(v * 0.70710678f));
      if (ACT == 2) v = fmaxf(v, 0.0f);
      if (W32) C32[(size_t)m * ldc + n] = v;
      if (W16) C16[(size_t)m * ldc + n] = (_Float16)v;
    }
  }
}

// =====================================================================
// LayerNorm with residual: out = LN(x + y) * g + b  (in-place safe on x32)
// =====================================================================
__global__ void k_ln(const float* __restrict__ X, const float* __restrict__ Y,
                     const float* __restrict__ g, const float* __restrict__ b,
                     float* __restrict__ O32, _Float16* __restrict__ O16) {
  const int row = blockIdx.x;
  const int t = threadIdx.x, c = t * 4;
  const float4 xv = *(const float4*)(X + (size_t)row * DIM + c);
  const float4 yv = *(const float4*)(Y + (size_t)row * DIM + c);
  float v[4] = {xv.x + yv.x, xv.y + yv.y, xv.z + yv.z, xv.w + yv.w};
  float s = v[0] + v[1] + v[2] + v[3];
  float q = v[0]*v[0] + v[1]*v[1] + v[2]*v[2] + v[3]*v[3];
#pragma unroll
  for (int off = 16; off >= 1; off >>= 1) { s += __shfl_xor(s, off, 32); q += __shfl_xor(q, off, 32); }
  __shared__ float ss[8], sq[8];
  const int wave = t >> 5, lane = t & 31;
  if (lane == 0) { ss[wave] = s; sq[wave] = q; }
  __syncthreads();
  if (t == 0) {
    float S = 0.f, Q = 0.f;
#pragma unroll
    for (int i = 0; i < 8; ++i) { S += ss[i]; Q += sq[i]; }
    const float mu = S * (1.0f / DIM);
    ss[0] = mu;
    sq[0] = rsqrtf(Q * (1.0f / DIM) - mu * mu + 1e-5f);
  }
  __syncthreads();
  const float mu = ss[0], rs = sq[0];
  float* o32 = O32 + (size_t)row * DIM + c;
  _Float16* o16 = O16 + (size_t)row * DIM + c;
#pragma unroll
  for (int i = 0; i < 4; ++i) {
    const float o = (v[i] - mu) * rs * g[c + i] + b[c + i];
    o32[i] = o; o16[i] = (_Float16)o;
  }
}

// =====================================================================
// Attention: one wave per (16-query tile, head).
// scores = Q Kt / 8 -> softmax -> P V, all via WMMA; writes o as f16.
// =====================================================================
__global__ void k_attn(const _Float16* __restrict__ qkv, _Float16* __restrict__ o16) {
  const int q0 = blockIdx.x * 16;
  const int h = blockIdx.y;
  const int lane = threadIdx.x & 31;
  const int hs = lane >> 4, l16 = lane & 15;

  __shared__ __align__(16) float sc[16][SEQ];       // 24 KB
  __shared__ __align__(16) _Float16 pm[16][SEQ];    // 12 KB

  // ---- phase 1: scores[16, 384] ----
  const _Float16* Arow = qkv + (size_t)(q0 + l16) * (3 * DIM) + h * HD;
  for (int n0 = 0; n0 < SEQ; n0 += 16) {
    const _Float16* Brow = qkv + (size_t)(n0 + l16) * (3 * DIM) + DIM + h * HD;
    v8f acc = {};
#pragma unroll
    for (int k = 0; k < HD; k += 32) {
      v16h a = load_a_tile(Arow, k, hs);
      v16h b = *(const v16h*)(Brow + k + hs * 16);
      acc = __builtin_amdgcn_wmma_f32_16x16x32_f16(false, a, false, b, (short)0, acc,
                                                   false, false);
    }
#pragma unroll
    for (int r = 0; r < 8; ++r) sc[r + hs * 8][n0 + l16] = acc[r] * 0.125f;
  }
  __syncthreads();

  // ---- phase 2: softmax (row = l16, each half-wave handles 192 cols) ----
  {
    const int row = l16, cb = hs * 192;
    float mx = -1e30f;
    for (int c = 0; c < 192; ++c) mx = fmaxf(mx, sc[row][cb + c]);
    mx = fmaxf(mx, __shfl_xor(mx, 16, 32));
    float sum = 0.f;
    for (int c = 0; c < 192; ++c) { float e = __expf(sc[row][cb + c] - mx); sc[row][cb + c] = e; sum += e; }
    sum += __shfl_xor(sum, 16, 32);
    const float inv = 1.0f / sum;
    for (int c = 0; c < 192; ++c) pm[row][cb + c] = (_Float16)(sc[row][cb + c] * inv);
  }
  __syncthreads();

  // ---- phase 3: out[16, 64] = P @ V ----
  for (int nt = 0; nt < 4; ++nt) {
    const int n0 = nt * 16;
    const _Float16* vbase = qkv + 2 * DIM + h * HD + n0 + l16;
    v8f acc = {};
    for (int k = 0; k < SEQ; k += 32) {
      v8h lo = *(const v8h*)(&pm[l16][k + hs * 16]);
      v8h hi = *(const v8h*)(&pm[l16][k + hs * 16 + 8]);
      v16h a;
#pragma unroll
      for (int t = 0; t < 8; ++t) { a[t] = lo[t]; a[8 + t] = hi[t]; }
      v16h b;
      const int kb = k + hs * 16;
#pragma unroll
      for (int t = 0; t < 16; ++t) b[t] = vbase[(size_t)(kb + t) * (3 * DIM)];
      acc = __builtin_amdgcn_wmma_f32_16x16x32_f16(false, a, false, b, (short)0, acc,
                                                   false, false);
    }
#pragma unroll
    for (int r = 0; r < 8; ++r)
      o16[(size_t)(q0 + r + hs * 8) * DIM + h * HD + n0 + l16] = (_Float16)acc[r];
  }
}

// =====================================================================
// Fused contact head. Block = 1 wave, grid (j-tile, i-tile, i-local).
// h[16,1024] f16 built in LDS (relu(ai[i]+aj[j]+cb1)); cw2 panels are
// double-buffered into LDS by the Tensor Data Mover (tensor_load_to_lds,
// s_wait_tensorcnt) while WMMA consumes the previous panel. relu+bias,
// dot with cw3, sigmoid -> 16 contact values. h never touches HBM.
// LDS: 32 KB (hA) + 2x16 KB (B panels) = 64 KB.
// =====================================================================
__global__ void k_contact(const float* __restrict__ ai, const float* __restrict__ aj,
                          const float* __restrict__ cb1,
                          const _Float16* __restrict__ cw2_16,
                          const float* __restrict__ cb2,
                          const float* __restrict__ cw3, const float* __restrict__ cb3,
                          float* __restrict__ out) {
  const int j0 = blockIdx.x * 16;
  const int i = blockIdx.y * 16 + blockIdx.z;
  const int lane = threadIdx.x & 31;
  const int hs = lane >> 4, l16 = lane & 15;

  __shared__ __align__(16) _Float16 hA[16][DIM];       // 32 KB
  __shared__ __align__(16) _Float16 Bb[2][16][512];    // 2 x 16 KB TDM buffers

  // stage h rows (row = j_local): two lanes per row, 512 cols each
  {
    const int row = lane >> 1;
    const int cb = (lane & 1) * 512;
    const float* air = ai + (size_t)i * DIM;
    const float* ajr = aj + (size_t)(j0 + row) * DIM;
    for (int c = cb; c < cb + 512; c += 4) {
      const float4 av = *(const float4*)(air + c);
      const float4 jv = *(const float4*)(ajr + c);
      const float4 bv = *(const float4*)(cb1 + c);
      hA[row][c + 0] = (_Float16)fmaxf(av.x + jv.x + bv.x, 0.0f);
      hA[row][c + 1] = (_Float16)fmaxf(av.y + jv.y + bv.y, 0.0f);
      hA[row][c + 2] = (_Float16)fmaxf(av.z + jv.z + bv.z, 0.0f);
      hA[row][c + 3] = (_Float16)fmaxf(av.w + jv.w + bv.w, 0.0f);
    }
  }
  __syncthreads();

  const unsigned ldsB[2] = {(unsigned)(size_t)(void*)&Bb[0][0][0],
                            (unsigned)(size_t)(void*)&Bb[1][0][0]};
  // 64 panels: panel p -> n0 = (p>>1)*16, k-half = (p&1)*512 (16 rows x 512 halfs)
  auto issue = [&](int p) {
    const int n0 = (p >> 1) * 16;
    const int kh = (p & 1) * 512;
    tdm_load_2d_f16(ldsB[p & 1], cw2_16 + (size_t)n0 * DIM + kh, 512, 16, DIM);
  };

  float csum[8];
#pragma unroll
  for (int r = 0; r < 8; ++r) csum[r] = 0.0f;

  issue(0);
  v8f acc = {};
  for (int p = 0; p < 64; ++p) {
    if (p + 1 < 64) {
      issue(p + 1);
      __builtin_amdgcn_s_wait_tensorcnt(1);   // panel p landed, p+1 in flight
    } else {
      __builtin_amdgcn_s_wait_tensorcnt(0);
    }
    const _Float16(*Bp)[512] = Bb[p & 1];
    const int kh = (p & 1) * 512;
    for (int kk = 0; kk < 512; kk += 32) {
      const int k = kh + kk;
      v8h lo = *(const v8h*)(&hA[l16][k + hs * 16]);
      v8h hi = *(const v8h*)(&hA[l16][k + hs * 16 + 8]);
      v16h a;
#pragma unroll
      for (int t = 0; t < 8; ++t) { a[t] = lo[t]; a[8 + t] = hi[t]; }
      v16h b = *(const v16h*)(&Bp[l16][kk + hs * 16]);
      acc = __builtin_amdgcn_wmma_f32_16x16x32_f16(false, a, false, b, (short)0, acc,
                                                   false, false);
    }
    if (p & 1) {  // finished full K for this n0 tile
      const int n0 = (p >> 1) * 16;
      const float w3 = cw3[n0 + l16];
      const float b2 = cb2[n0 + l16];
#pragma unroll
      for (int r = 0; r < 8; ++r) {
        csum[r] += fmaxf(acc[r] + b2, 0.0f) * w3;
        acc[r] = 0.0f;
      }
    }
  }

  // reduce across the 16 lanes of each half-wave (same m, all n)
#pragma unroll
  for (int r = 0; r < 8; ++r) {
#pragma unroll
    for (int off = 1; off < 16; off <<= 1) csum[r] += __shfl_xor(csum[r], off, 32);
  }
  if (l16 == 0) {
    const float b3 = cb3[0];
#pragma unroll
    for (int r = 0; r < 8; ++r) {
      const int j = j0 + r + hs * 8;
      const float v = csum[r] + b3;
      out[(size_t)i * SEQ + j] = 1.0f / (1.0f + __expf(-v));
    }
  }
}

// =====================================================================
// Secondary head stage 2: out[s, o] = t[s,:] . sw2[o,:] + sb2[o]  (N=8)
// =====================================================================
__global__ void k_sec2(const float* __restrict__ t32, const float* __restrict__ sw2,
                       const float* __restrict__ sb2, float* __restrict__ out) {
  const int s = blockIdx.x;
  const int o = threadIdx.x >> 5;
  const int lane = threadIdx.x & 31;
  const float* tr = t32 + (size_t)s * 512;
  const float* wr = sw2 + (size_t)o * 512;
  float sum = 0.f;
  for (int c = lane; c < 512; c += 32) sum += tr[c] * wr[c];
#pragma unroll
  for (int off = 16; off >= 1; off >>= 1) sum += __shfl_xor(sum, off, 32);
  if (lane == 0) out[(size_t)s * 8 + o] = sum + sb2[o];
}

// =====================================================================
// Host orchestration
// =====================================================================
static inline size_t align256(size_t x) { return (x + 255) & ~(size_t)255; }

extern "C" void kernel_launch(void* const* d_in, const int* in_sizes, int n_in,
                              void* d_out, int out_size, void* d_ws, size_t ws_size,
                              hipStream_t stream) {
  const int*   seq    = (const int*)  d_in[0];
  const float* emb    = (const float*)d_in[1];
  const float* pos    = (const float*)d_in[2];
  const float* qkv_w  = (const float*)d_in[3];
  const float* qkv_b  = (const float*)d_in[4];
  const float* out_w  = (const float*)d_in[5];
  const float* out_b  = (const float*)d_in[6];
  const float* ln1_s  = (const float*)d_in[7];
  const float* ln1_b  = (const float*)d_in[8];
  const float* ffn_w1 = (const float*)d_in[9];
  const float* ffn_b1 = (const float*)d_in[10];
  const float* ffn_w2 = (const float*)d_in[11];
  const float* ffn_b2 = (const float*)d_in[12];
  const float* ln2_s  = (const float*)d_in[13];
  const float* ln2_b  = (const float*)d_in[14];
  const float* cw1    = (const float*)d_in[15];
  const float* cb1    = (const float*)d_in[16];
  const float* cw2    = (const float*)d_in[17];
  const float* cb2    = (const float*)d_in[18];
  const float* cw3    = (const float*)d_in[19];
  const float* cb3    = (const float*)d_in[20];
  const float* sw1    = (const float*)d_in[21];
  const float* sb1    = (const float*)d_in[22];
  const float* sw2    = (const float*)d_in[23];
  const float* sb2    = (const float*)d_in[24];

  float* outp = (float*)d_out;

  char* ws = (char*)d_ws;
  size_t off = 0;
  auto alloc = [&](size_t bytes) -> void* {
    void* p = ws + off; off = align256(off + bytes); return p;
  };
  float*     x32   = (float*)    alloc((size_t)SEQ * DIM * 4);
  _Float16*  x16   = (_Float16*) alloc((size_t)SEQ * DIM * 2);
  _Float16*  qkv16 = (_Float16*) alloc((size_t)SEQ * 3 * DIM * 2);
  _Float16*  o16   = (_Float16*) alloc((size_t)SEQ * DIM * 2);
  float*     y32   = (float*)    alloc((size_t)SEQ * DIM * 4);
  _Float16*  t16   = (_Float16*) alloc((size_t)SEQ * DFF * 2);
  float*     t32   = (float*)    alloc((size_t)SEQ * 512 * 4);
  float*     ai32  = (float*)    alloc((size_t)SEQ * DIM * 4);
  float*     aj32  = (float*)    alloc((size_t)SEQ * DIM * 4);
  _Float16*  wf16  = (_Float16*) alloc((size_t)DFF * DIM * 2);  // weight slab (max 8 MB)

  auto cvt = [&](const float* src, _Float16* dst, int n) {
    k_cvt<<<dim3((n + 1023) / 1024), dim3(256), 0, stream>>>(src, dst, n);
  };

  // ---- embedding ----
  k_embed<<<dim3(SEQ), dim3(256), 0, stream>>>(seq, emb, pos, x32, x16);

  // ---- encoder layers ----
  for (int l = 0; l < NL; ++l) {
    // qkv = x @ qkv_w[l].T + qkv_b[l]  (f16 out for attention)
    cvt(qkv_w + (size_t)l * 3 * DIM * DIM, wf16, 3 * DIM * DIM);
    k_gemm_nt4<0, false, true><<<dim3(3 * DIM / 64, SEQ / 16), dim3(32), 0, stream>>>(
        x16, DIM, wf16, DIM, qkv_b + (size_t)l * 3 * DIM,
        nullptr, qkv16, 3 * DIM, SEQ, 3 * DIM, DIM);

    // attention -> o16
    k_attn<<<dim3(SEQ / 16, NH), dim3(32), 0, stream>>>(qkv16, o16);

    // y = o @ out_w[l].T + out_b[l]
    cvt(out_w + (size_t)l * DIM * DIM, wf16, DIM * DIM);
    k_gemm_nt4<0, true, false><<<dim3(DIM / 64, SEQ / 16), dim3(32), 0, stream>>>(
        o16, DIM, wf16, DIM, out_b + (size_t)l * DIM,
        y32, nullptr, DIM, SEQ, DIM, DIM);

    // x = LN(x + y)
    k_ln<<<dim3(SEQ), dim3(256), 0, stream>>>(x32, y32, ln1_s + (size_t)l * DIM,
                                              ln1_b + (size_t)l * DIM, x32, x16);

    // t = gelu(x @ ffn_w1[l].T + b1)
    cvt(ffn_w1 + (size_t)l * DFF * DIM, wf16, DFF * DIM);
    k_gemm_nt4<1, false, true><<<dim3(DFF / 64, SEQ / 16), dim3(32), 0, stream>>>(
        x16, DIM, wf16, DIM, ffn_b1 + (size_t)l * DFF,
        nullptr, t16, DFF, SEQ, DFF, DIM);

    // y = t @ ffn_w2[l].T + b2
    cvt(ffn_w2 + (size_t)l * DIM * DFF, wf16, DIM * DFF);
    k_gemm_nt4<0, true, false><<<dim3(DIM / 64, SEQ / 16), dim3(32), 0, stream>>>(
        t16, DFF, wf16, DFF, ffn_b2 + (size_t)l * DIM,
        y32, nullptr, DIM, SEQ, DIM, DFF);

    // x = LN(x + y)
    k_ln<<<dim3(SEQ), dim3(256), 0, stream>>>(x32, y32, ln2_s + (size_t)l * DIM,
                                              ln2_b + (size_t)l * DIM, x32, x16);
  }

  // ---- contact head ----
  // ai = enc @ cw1[:, :D].T ; aj = enc @ cw1[:, D:].T   (cw1 is [D, 2D] row-major)
  cvt(cw1, wf16, DIM * 2 * DIM);
  k_gemm_nt4<0, true, false><<<dim3(DIM / 64, SEQ / 16), dim3(32), 0, stream>>>(
      x16, DIM, wf16, 2 * DIM, nullptr, ai32, nullptr, DIM, SEQ, DIM, DIM);
  k_gemm_nt4<0, true, false><<<dim3(DIM / 64, SEQ / 16), dim3(32), 0, stream>>>(
      x16, DIM, wf16 + DIM, 2 * DIM, nullptr, aj32, nullptr, DIM, SEQ, DIM, DIM);

  // fused pairwise head (cw2 as f16; reuses wf16 slab); TDM streams B panels
  cvt(cw2, wf16, 512 * DIM);
  k_contact<<<dim3(SEQ / 16, SEQ / 16, 16), dim3(32), 0, stream>>>(
      ai32, aj32, cb1, wf16, cb2, cw3, cb3, outp);

  // ---- secondary head ----
  cvt(sw1, wf16, 512 * DIM);
  k_gemm_nt4<2, true, false><<<dim3(512 / 64, SEQ / 16), dim3(32), 0, stream>>>(
      x16, DIM, wf16, DIM, sb1, t32, nullptr, 512, SEQ, 512, DIM);
  k_sec2<<<dim3(SEQ), dim3(256), 0, stream>>>(t32, sw2, sb2,
                                              outp + (size_t)SEQ * SEQ);
}
